// TransformerGNN_19937238188414
// MI455X (gfx1250) — matrix operand order repaired and checked by reading, not compiled, not run on hardware
//
#include <hip/hip_runtime.h>
#include <hip/hip_bf16.h>

typedef __attribute__((ext_vector_type(16))) _Float16 v16h;
typedef __attribute__((ext_vector_type(8)))  float    v8f;

// ---------------------------------------------------------------------------
// Weight packing: put [K x Nout] fp32 weights into the CDNA5 WMMA B-operand
// register image for v_wmma_f32_16x16x32_f16:
//   lane l (0..15):  N = l,    holds K = ktile*32 + {0..7,16..23} (f16 j in that order)
//   lane l (16..31): N = l-16, holds K = ktile*32 + {8..15,24..31}
// Stored as [tile][lane][16 f16] so the GEMM does one 32B load per lane/tile.
// ---------------------------------------------------------------------------
__global__ void pack_l1(const float* __restrict__ Wq, const float* __restrict__ Wk,
                        const float* __restrict__ Wv, const float* __restrict__ Ws,
                        const float* __restrict__ bq, const float* __restrict__ bk,
                        const float* __restrict__ bv, const float* __restrict__ bs,
                        _Float16* __restrict__ wpack, float* __restrict__ bias) {
  int t = blockIdx.x * blockDim.x + threadIdx.x;
  if (t < 16 * 512) {                     // 4 N-tiles x 4 K-tiles x 32 lanes x 16
    int tile = t >> 9;                    // 0..15
    int nt = tile >> 2, kt = tile & 3;
    int lane = (t >> 4) & 31;
    int j = t & 15;
    int n = lane & 15;
    int k = kt * 32 + ((j >> 3) * 16) + ((lane >> 4) * 8) + (j & 7);   // 0..127
    const float* W = (nt == 0) ? Wq : (nt == 1) ? Wk : (nt == 2) ? Wv : Ws;
    wpack[t] = (_Float16)W[k * 16 + n];         // W is [128 x 16] row-major
  }
  if (t < 64) {
    int nt = t >> 4, n = t & 15;
    const float* B = (nt == 0) ? bq : (nt == 1) ? bk : (nt == 2) ? bv : bs;
    bias[t] = B[n];
  }
}

// Layer 2: only q2/k2 are needed (layer-2 'out' is discarded by the reference).
// K=16 real, zero-padded to the 32-wide WMMA K tile.
__global__ void pack_l2(const float* __restrict__ Wq, const float* __restrict__ Wk,
                        const float* __restrict__ bq, const float* __restrict__ bk,
                        _Float16* __restrict__ wpack, float* __restrict__ bias) {
  int t = blockIdx.x * blockDim.x + threadIdx.x;
  if (t < 4 * 512) {                      // 4 N-tiles (q:0..31, k:32..63), 1 K-tile
    int tile = t >> 9;                    // 0..3
    int lane = (t >> 4) & 31;
    int j = t & 15;
    int n = lane & 15;
    int kk = (j >> 3) * 16 + (lane >> 4) * 8 + (j & 7);   // 0..31 (>=16 -> zero pad)
    int c = tile * 16 + n;                // 0..63 concat column
    const float* W = (c < 32) ? Wq : Wk;  // W is [16 x 32] row-major
    int coln = c & 31;
    float v = (kk < 16) ? W[kk * 32 + coln] : 0.0f;
    wpack[t] = (_Float16)v;
  }
  if (t < 64) {
    const float* B = (t < 32) ? bq : bk;
    bias[t] = B[t & 31];
  }
}

// ---------------------------------------------------------------------------
// Node GEMM, layer 1: [N x 128] @ [128 x 64(concat q|k|v|s)] via WMMA.
// One wave handles a 16-row M block; 4 K-steps x 4 N-tiles = 16 WMMAs.
// A-operand ISA layout: lane = M (mod 16); lanes 0-15 carry K {0..7,16..23}
// of each 32-K tile, lanes 16-31 carry K {8..15,24..31}.
// ---------------------------------------------------------------------------
__device__ inline v16h make_a(const float* xr) {
  float4 a0 = *(const float4*)(xr + 0);
  float4 a1 = *(const float4*)(xr + 4);
  float4 b0 = *(const float4*)(xr + 16);
  float4 b1 = *(const float4*)(xr + 20);
  v16h A;
  A[0] = (_Float16)a0.x;  A[1] = (_Float16)a0.y;  A[2] = (_Float16)a0.z;  A[3] = (_Float16)a0.w;
  A[4] = (_Float16)a1.x;  A[5] = (_Float16)a1.y;  A[6] = (_Float16)a1.z;  A[7] = (_Float16)a1.w;
  A[8] = (_Float16)b0.x;  A[9] = (_Float16)b0.y;  A[10] = (_Float16)b0.z; A[11] = (_Float16)b0.w;
  A[12] = (_Float16)b1.x; A[13] = (_Float16)b1.y; A[14] = (_Float16)b1.z; A[15] = (_Float16)b1.w;
  return A;
}

__global__ __launch_bounds__(256) void gemm_nodes1(
    const float* __restrict__ x, const _Float16* __restrict__ wpack,
    const float* __restrict__ bias, float* __restrict__ out, int nBlocks) {
  int w = blockIdx.x * (blockDim.x >> 5) + (threadIdx.x >> 5);
  int lane = threadIdx.x & 31;
  if (w >= nBlocks) return;
  int row0 = w * 16;
  int m = lane & 15;
  int koff = (lane >> 4) * 8;
  const v16h* wp = reinterpret_cast<const v16h*>(wpack);
  v8f acc[4] = {v8f{}, v8f{}, v8f{}, v8f{}};
#pragma unroll
  for (int kt = 0; kt < 4; ++kt) {
    const float* xr = x + (long long)(row0 + m) * 128 + kt * 32 + koff;
    v16h A = make_a(xr);
#pragma unroll
    for (int nt = 0; nt < 4; ++nt) {
      acc[nt] = __builtin_amdgcn_wmma_f32_16x16x32_f16(
          false, A, false, wp[(nt * 4 + kt) * 32 + lane], (short)0, acc[nt], false, false);
    }
  }
  int nn = lane & 15;
  int rsel = (lane >> 4) * 8;     // C/D layout: vgpr j -> M = j + 8*lane[4]
#pragma unroll
  for (int nt = 0; nt < 4; ++nt) {
    float b = bias[nt * 16 + nn];
#pragma unroll
    for (int j = 0; j < 8; ++j) {
      out[(long long)(row0 + j + rsel) * 64 + nt * 16 + nn] = acc[nt][j] + b;
    }
  }
}

// Layer-2 node GEMM: h [N x 16] @ [16 x 64(q2|k2)], single zero-padded K tile.
__global__ __launch_bounds__(256) void gemm_nodes2(
    const float* __restrict__ h, const _Float16* __restrict__ wpack,
    const float* __restrict__ bias, float* __restrict__ out, int nBlocks) {
  int w = blockIdx.x * (blockDim.x >> 5) + (threadIdx.x >> 5);
  int lane = threadIdx.x & 31;
  if (w >= nBlocks) return;
  int row0 = w * 16;
  int m = lane & 15;
  int koff = (lane >> 4) * 8;     // lanes 0-15: K 0..7; lanes 16-31: K 8..15
  const float* hr = h + (long long)(row0 + m) * 16 + koff;
  float4 a0 = *(const float4*)(hr + 0);
  float4 a1 = *(const float4*)(hr + 4);
  v16h A;
  A[0] = (_Float16)a0.x; A[1] = (_Float16)a0.y; A[2] = (_Float16)a0.z; A[3] = (_Float16)a0.w;
  A[4] = (_Float16)a1.x; A[5] = (_Float16)a1.y; A[6] = (_Float16)a1.z; A[7] = (_Float16)a1.w;
#pragma unroll
  for (int j = 8; j < 16; ++j) A[j] = (_Float16)0.0f;   // zero-pad K 16..31
  const v16h* wp = reinterpret_cast<const v16h*>(wpack);
  v8f acc[4] = {v8f{}, v8f{}, v8f{}, v8f{}};
#pragma unroll
  for (int nt = 0; nt < 4; ++nt) {
    acc[nt] = __builtin_amdgcn_wmma_f32_16x16x32_f16(
        false, A, false, wp[nt * 32 + lane], (short)0, acc[nt], false, false);
  }
  int nn = lane & 15;
  int rsel = (lane >> 4) * 8;
#pragma unroll
  for (int nt = 0; nt < 4; ++nt) {
    float b = bias[nt * 16 + nn];
#pragma unroll
    for (int j = 0; j < 8; ++j) {
      out[(long long)(row0 + j + rsel) * 64 + nt * 16 + nn] = acc[nt][j] + b;
    }
  }
}

// ---------------------------------------------------------------------------
// Edge-side kernels (HBM-streaming, L2-resident gathers, L2 float atomics)
// ---------------------------------------------------------------------------
__global__ void init_nodes(float* __restrict__ nodeMax, float* __restrict__ nodeSum, int N) {
  int i = blockIdx.x * blockDim.x + threadIdx.x;
  if (i < N) { nodeMax[i] = -3.402823466e38f; nodeSum[i] = 0.0f; }
}

__global__ void zero_f32(float* __restrict__ p, int n) {
  int i = blockIdx.x * blockDim.x + threadIdx.x;
  if (i < n) p[i] = 0.0f;
}

template <int C>
__global__ void edge_logits(const int* __restrict__ src, const int* __restrict__ dst,
                            const float* __restrict__ ea, const float* __restrict__ We,
                            const float* __restrict__ qkvs, int qoff, int koff,
                            float* __restrict__ logits, float* __restrict__ nodeMax,
                            float scale, int E) {
  int e = blockIdx.x * blockDim.x + threadIdx.x;
  if (e >= E) return;
  int s = src[e], d = dst[e];
  float w = ea[e];
  const float4* qp = (const float4*)(qkvs + (long long)d * 64 + qoff);
  const float4* kp = (const float4*)(qkvs + (long long)s * 64 + koff);
  const float4* wp = (const float4*)We;
  float acc = 0.0f;
#pragma unroll
  for (int i = 0; i < C / 4; ++i) {
    float4 q = qp[i], k = kp[i], ww = wp[i];
    acc += q.x * (k.x + w * ww.x) + q.y * (k.y + w * ww.y) +
           q.z * (k.z + w * ww.z) + q.w * (k.w + w * ww.w);
  }
  float lg = acc * scale;
  logits[e] = lg;
  __hip_atomic_fetch_max(&nodeMax[d], lg, __ATOMIC_RELAXED, __HIP_MEMORY_SCOPE_AGENT);
}

__global__ void edge_exp(const float* __restrict__ logits, const float* __restrict__ nodeMax,
                         const int* __restrict__ dst, float* __restrict__ aRaw,
                         float* __restrict__ nodeSum, int E) {
  int e = blockIdx.x * blockDim.x + threadIdx.x;
  if (e >= E) return;
  int d = dst[e];
  float a = __expf(logits[e] - nodeMax[d]);
  aRaw[e] = a;
  atomicAdd(&nodeSum[d], a);
}

// Layer 1: normalize alpha in place and scatter-add alpha*(v[src]+edge_vec).
__global__ void edge_norm_agg1(float* __restrict__ alpha, const float* __restrict__ nodeSum,
                               const int* __restrict__ src, const int* __restrict__ dst,
                               const float* __restrict__ ea, const float* __restrict__ We,
                               const float* __restrict__ qkvs, float* __restrict__ agg, int E) {
  int e = blockIdx.x * blockDim.x + threadIdx.x;
  if (e >= E) return;
  int s = src[e], d = dst[e];
  float al = alpha[e] / (nodeSum[d] + 1e-16f);
  alpha[e] = al;
  float w = ea[e];
  const float4* vp = (const float4*)(qkvs + (long long)s * 64 + 32);
  const float4* wp = (const float4*)We;
  float* ag = agg + (long long)d * 16;
#pragma unroll
  for (int i = 0; i < 4; ++i) {
    float4 v = vp[i], ww = wp[i];
    atomicAdd(ag + i * 4 + 0, al * (v.x + w * ww.x));
    atomicAdd(ag + i * 4 + 1, al * (v.y + w * ww.y));
    atomicAdd(ag + i * 4 + 2, al * (v.z + w * ww.z));
    atomicAdd(ag + i * 4 + 3, al * (v.w + w * ww.w));
  }
}

// h = relu(agg + skip), in place over agg.
__global__ void relu_skip(float* __restrict__ agg, const float* __restrict__ qkvs1, int n16) {
  int t = blockIdx.x * blockDim.x + threadIdx.x;
  if (t >= n16) return;
  int node = t >> 4, n = t & 15;
  float v = agg[t] + qkvs1[(long long)node * 64 + 48 + n];
  agg[t] = fmaxf(v, 0.0f);
}

// Layer 2: normalize and write alpha2 straight to the output buffer.
__global__ void edge_norm_out2(const float* __restrict__ aRaw, const float* __restrict__ nodeSum,
                               const int* __restrict__ dst, float* __restrict__ out, int E) {
  int e = blockIdx.x * blockDim.x + threadIdx.x;
  if (e >= E) return;
  out[e] = aRaw[e] / (nodeSum[dst[e]] + 1e-16f);
}

__global__ void copy_edges(const int* __restrict__ edges, int* __restrict__ out, int n) {
  int t = blockIdx.x * blockDim.x + threadIdx.x;
  if (t < n) out[t] = edges[t];
}

// ---------------------------------------------------------------------------
extern "C" void kernel_launch(void* const* d_in, const int* in_sizes, int n_in,
                              void* d_out, int out_size, void* d_ws, size_t ws_size,
                              hipStream_t stream) {
  const float* x     = (const float*)d_in[0];
  const int*   edges = (const int*)d_in[1];
  const float* ew    = (const float*)d_in[2];
  const float* Wq1 = (const float*)d_in[3];  const float* bq1 = (const float*)d_in[4];
  const float* Wk1 = (const float*)d_in[5];  const float* bk1 = (const float*)d_in[6];
  const float* Wv1 = (const float*)d_in[7];  const float* bv1 = (const float*)d_in[8];
  const float* We1 = (const float*)d_in[9];
  const float* Ws1 = (const float*)d_in[10]; const float* bs1 = (const float*)d_in[11];
  const float* Wq2 = (const float*)d_in[12]; const float* bq2 = (const float*)d_in[13];
  const float* Wk2 = (const float*)d_in[14]; const float* bk2 = (const float*)d_in[15];
  // Wv2/bv2/We2? -> We2 IS needed (edge embedding feeds logits); Wv2/Ws2 are not
  const float* We2 = (const float*)d_in[18];
  (void)d_in[16]; (void)d_in[17]; (void)d_in[19]; (void)d_in[20];
  (void)n_in; (void)out_size; (void)ws_size;

  const int N = in_sizes[0] / 128;
  const int E = in_sizes[1] / 2;
  const int* src = edges;
  const int* dst = edges + E;

  // Workspace layout (fp32 words, then f16 packs; ~84 MB for N=100K, E=3.2M)
  float* f = (float*)d_ws;
  float* qkvs1   = f; f += (size_t)N * 64;   // q|k|v|skip  layer 1
  float* qkvs2   = f; f += (size_t)N * 64;   // q2|k2       layer 2
  float* agg     = f; f += (size_t)N * 16;   // agg, then h (in place)
  float* nodeMax = f; f += N;
  float* nodeSum = f; f += N;
  float* logits  = f; f += E;                // also layer-2 raw exp (in place)
  float* alpha1  = f; f += E;
  float* bias1   = f; f += 64;
  float* bias2   = f; f += 64;
  _Float16* wpack1 = (_Float16*)f;           // 16 tiles * 512 f16
  _Float16* wpack2 = wpack1 + 16 * 512;      //  4 tiles * 512 f16

  const int T = 256;
  const int nBlocks = (N + 15) / 16;
  const int gemmGrid = (nBlocks * 32 + T - 1) / T;
  const int eGrid = (E + T - 1) / T;
  const int nGrid = (N + T - 1) / T;
  const int n16Grid = (N * 16 + T - 1) / T;

  // --- pack weights into WMMA B layout ---
  pack_l1<<<(16 * 512 + T - 1) / T, T, 0, stream>>>(Wq1, Wk1, Wv1, Ws1, bq1, bk1, bv1, bs1,
                                                    wpack1, bias1);
  pack_l2<<<(4 * 512 + T - 1) / T, T, 0, stream>>>(Wq2, Wk2, bq2, bk2, wpack2, bias2);

  // --- layer 1 ---
  gemm_nodes1<<<gemmGrid, T, 0, stream>>>(x, wpack1, bias1, qkvs1, nBlocks);
  init_nodes<<<nGrid, T, 0, stream>>>(nodeMax, nodeSum, N);
  zero_f32<<<n16Grid, T, 0, stream>>>(agg, N * 16);
  edge_logits<16><<<eGrid, T, 0, stream>>>(src, dst, ew, We1, qkvs1, 0, 16,
                                           logits, nodeMax, 0.25f, E);
  edge_exp<<<eGrid, T, 0, stream>>>(logits, nodeMax, dst, alpha1, nodeSum, E);
  edge_norm_agg1<<<eGrid, T, 0, stream>>>(alpha1, nodeSum, src, dst, ew, We1, qkvs1, agg, E);
  relu_skip<<<n16Grid, T, 0, stream>>>(agg, qkvs1, N * 16);

  // --- layer 2 (only alpha2 is needed) ---
  gemm_nodes2<<<gemmGrid, T, 0, stream>>>(agg, wpack2, bias2, qkvs2, nBlocks);
  init_nodes<<<nGrid, T, 0, stream>>>(nodeMax, nodeSum, N);
  edge_logits<32><<<eGrid, T, 0, stream>>>(src, dst, alpha1, We2, qkvs2, 0, 32,
                                           logits, nodeMax, 0.17677669529663687f, E);
  edge_exp<<<eGrid, T, 0, stream>>>(logits, nodeMax, dst, logits, nodeSum, E);

  // Output: tuple (edges, alpha2[:,None]) concatenated flat in return order.
  copy_edges<<<(2 * E + T - 1) / T, T, 0, stream>>>(edges, (int*)d_out, 2 * E);
  edge_norm_out2<<<eGrid, T, 0, stream>>>(logits, nodeSum, dst, (float*)d_out + 2 * (size_t)E, E);
}
